// grouped_mapping_framework_36129264894589
// MI455X (gfx1250) — matrix operand order repaired and checked by reading, not compiled, not run on hardware
//
#include <hip/hip_runtime.h>
#include <math.h>
#include <stdint.h>

// ---------------- WMMA types & fragment helpers (CDNA5 / gfx1250, wave32) ---
typedef __attribute__((ext_vector_type(16))) _Float16 v16h;
typedef __attribute__((ext_vector_type(8)))  float    v8f;
typedef unsigned int u32;
typedef u32 uv4 __attribute__((ext_vector_type(4)));
typedef int  iv4 __attribute__((ext_vector_type(4)));
typedef int  iv8 __attribute__((ext_vector_type(8)));

union FragU {
  v16h v;
  uint4 q[2];
};

// A matrix 16x32 f16 fragment, from row-major [row*stride + k] halves.
// lanes 0-15: row M=lane, K={k0..k0+7, k0+16..k0+23};
// lanes 16-31: row M=lane-16, K={k0+8..k0+15, k0+24..k0+31}.
// stride, k0 multiples of 8 (16B alignment).
__device__ __forceinline__ v16h load_frag_a(const _Float16* __restrict__ base,
                                            int stride, int m0, int k0, int lane) {
  int r  = lane & 15;
  int kb = k0 + ((lane & 16) ? 8 : 0);
  const _Float16* p = base + (m0 + r) * stride + kb;
  FragU f;
  f.q[0] = *(const uint4*)(p);
  f.q[1] = *(const uint4*)(p + 16);
  return f.v;
}

// B matrix 32x16 f16 fragment from TRANSPOSED buffer [col*stride + k].
// lanes 0-15: col N=lane, K={k0..k0+7, k0+16..k0+23}; lanes 16-31: col N=lane-16, K={k0+8..k0+15, k0+24..k0+31}.
__device__ __forceinline__ v16h load_frag_b(const _Float16* __restrict__ baseT,
                                            int stride, int k0, int n0, int lane) {
  int c  = n0 + (lane & 15);
  int kb = k0 + ((lane & 16) ? 8 : 0);
  const _Float16* p = baseT + c * stride + kb;
  FragU f;
  f.q[0] = *(const uint4*)(p);
  f.q[1] = *(const uint4*)(p + 16);
  return f.v;
}

__device__ __forceinline__ v8f wmma16(v16h a, v16h b, v8f c) {
  return __builtin_amdgcn_wmma_f32_16x16x32_f16(false, a, false, b, (short)0, c,
                                                false, false);
}

// ---------------- TDM: 2D f16 tile load Global->LDS with row padding --------
#if defined(__has_builtin)
#if __has_builtin(__builtin_amdgcn_tensor_load_to_lds) && \
    __has_builtin(__builtin_amdgcn_s_wait_tensorcnt)
#define HAVE_TDM 1
#endif
#endif

#ifdef HAVE_TDM
// Loads tile_h rows x tile_w halves from gsrc (row stride = stride_elems
// halves) into LDS at lds_dst, inserting pad_dwords of LDS padding every
// int_dwords DWORDs (codes per D# group1). data_size = 2 bytes.
__device__ __forceinline__ void tdm_load_2d(const _Float16* gsrc, void* lds_dst,
                                            u32 dim0, u32 dim1, u32 stride0,
                                            u32 tile_w, u32 tile_h,
                                            u32 pad_int_code, u32 pad_amt_code) {
  unsigned long long ga = (unsigned long long)(uintptr_t)gsrc;
  u32 lds_off = (u32)(uintptr_t)lds_dst;  // generic addr low 32b = LDS offset
  uv4 g0;
  g0.x = 1u;                                 // count=1, user descriptor
  g0.y = lds_off;                            // lds_addr (bytes)
  g0.z = (u32)(ga & 0xffffffffu);            // global_addr[31:0]
  g0.w = ((u32)(ga >> 32) & 0x01ffffffu) | (2u << 30);  // addr[56:32] | type=2
  iv8 g1;
  g1[0] = (int)((1u << 16) |                 // data_size = 1 (2 bytes)
                (1u << 20) |                 // pad_enable
                (pad_int_code << 22) | (pad_amt_code << 25));
  g1[1] = (int)((dim0 & 0xffffu) << 16);                       // dim0[15:0]
  g1[2] = (int)(((dim0 >> 16) & 0xffffu) | ((dim1 & 0xffffu) << 16));
  g1[3] = (int)(((dim1 >> 16) & 0xffffu) | ((tile_w & 0xffffu) << 16));
  g1[4] = (int)(tile_h & 0xffffu);          // tile_dim1 (tile_dim2 = 0)
  g1[5] = (int)stride0;                     // tensor_dim0_stride[31:0]
  g1[6] = 0;
  g1[7] = 0;
  iv4 z4 = {0, 0, 0, 0};
#if defined(__clang_major__) && (__clang_major__ >= 23)
  iv8 z8 = {0, 0, 0, 0, 0, 0, 0, 0};
  __builtin_amdgcn_tensor_load_to_lds(g0, g1, z4, z4, z8, 0);
#else
  __builtin_amdgcn_tensor_load_to_lds(g0, g1, z4, z4, 0);
#endif
}
#endif

// Stage a [rows x 64] f16 weight tile into LDS with stride-72 rows.
// TDM path: issued by wave 0, overlapped with caller's other staging;
// caller must call weight_stage_wait() then __syncthreads().
__device__ __forceinline__ void weight_stage_issue(const _Float16* __restrict__ g,
                                                   u32 row_stride, _Float16* sW,
                                                   int tid) {
#ifdef HAVE_TDM
  if (tid < 32)
    tdm_load_2d(g, sW, row_stride, 192, row_stride, 64, 192, /*32 dw*/ 4,
                /*4 dw*/ 3);
#else
  for (int i = tid; i < 192 * 64; i += 256) {
    int o = i >> 6, c = i & 63;
    sW[o * 72 + c] = g[o * row_stride + c];
  }
#endif
}

__device__ __forceinline__ void weight_stage_wait(int tid) {
#ifdef HAVE_TDM
  if (tid < 32) __builtin_amdgcn_s_wait_tensorcnt(0);
#endif
}

// ---------------- Problem constants ----------------------------------------
#define NB    128
#define CIN   64
#define TT    64
#define VV    25
#define KK    3
#define COUT  192      // == MID*K
#define COLS  1600     // T*V
#define XSTR  102400   // CIN*COLS
#define OSTR  307200   // COUT*COLS
#define EPSB  1e-5f

// ---------------- Kernel 0a: x f32 -> f16 (4 elems/thread) -----------------
__global__ void __launch_bounds__(256)
k_cvt_x(const float* __restrict__ x, _Float16* __restrict__ x16) {
  int gid = blockIdx.x * 256 + threadIdx.x;
  float4 v = ((const float4*)x)[gid];
  union { ushort4 u; _Float16 h[4]; } t;
  t.h[0] = (_Float16)v.x;
  t.h[1] = (_Float16)v.y;
  t.h[2] = (_Float16)v.z;
  t.h[3] = (_Float16)v.w;
  ((ushort4*)x16)[gid] = t.u;
}

// ---------------- Kernel 0b: weights f32 -> f16 ----------------------------
// layout: [0,12288) W_pre | [12288,49152) W_post | [49152,61440) W_down
__global__ void __launch_bounds__(256)
k_cvt_w(const float* __restrict__ Wpre, const float* __restrict__ Wpost,
        const float* __restrict__ Wdown, _Float16* __restrict__ w16) {
  int i = blockIdx.x * 256 + threadIdx.x;
  float v;
  if (i < 12288) v = Wpre[i];
  else if (i < 49152) v = Wpost[i - 12288];
  else v = Wdown[i - 49152];
  w16[i] = (_Float16)v;
}

// ---------------- Kernel 1: tmp[n,c,v] = mean_t x[n,c,t,v] -----------------
__global__ void __launch_bounds__(32)
k_mean(const float* __restrict__ x, float* __restrict__ tmp) {
  int c = blockIdx.x, n = blockIdx.y;
  int v = threadIdx.x;
  if (v < VV) {
    const float* p = x + (size_t)n * XSTR + c * COLS + v;
    float s = 0.f;
#pragma unroll 8
    for (int t = 0; t < TT; ++t) s += p[t * VV];
    tmp[n * (CIN * VV) + c * VV + v] = s * (1.f / 64.f);
  }
}

// ---------------- Kernel 2: build Adyn^T (padded 32x32 f16 per (n,o)) ------
__global__ void __launch_bounds__(256)
k_adyn(const float* __restrict__ tmp, const float* __restrict__ A,
       const float* __restrict__ alpha, const float* __restrict__ beta,
       const float* __restrict__ W1, const float* __restrict__ b1,
       const float* __restrict__ W2, const float* __restrict__ b2,
       _Float16* __restrict__ adyn) {
  __shared__ float sT[CIN * VV];
  __shared__ float sX1[64 * VV], sX2[64 * VV];
  __shared__ float sS[VV * VV];
  int tid = threadIdx.x;
  int k = blockIdx.x, n = blockIdx.y;

  for (int i = tid; i < CIN * VV; i += 256) sT[i] = tmp[n * (CIN * VV) + i];
  __syncthreads();

  for (int i = tid; i < 64 * VV; i += 256) {
    int c = i / VV, v = i - c * VV;
    int o = k * 64 + c;
    float s1 = b1[o], s2 = b2[o];
    const float* w1 = W1 + o * CIN;
    const float* w2 = W2 + o * CIN;
#pragma unroll 8
    for (int ci = 0; ci < CIN; ++ci) {
      float tv = sT[ci * VV + v];
      s1 += w1[ci] * tv;
      s2 += w2[ci] * tv;
    }
    sX1[i] = s1;
    sX2[i] = s2;
  }
  __syncthreads();

  for (int i = tid; i < VV * VV; i += 256) {
    int v = i / VV, w = i - v * VV;
    float s = 0.f;
#pragma unroll 8
    for (int c = 0; c < 64; ++c) s += sX1[c * VV + v] * sX2[c * VV + w];
    sS[i] = s;
  }
  __syncthreads();

  float beta0 = beta[0];
  if (tid < VV) {  // softmax over v for column w=tid, scaled by beta0
    int w = tid;
    float mx = -1e30f;
    for (int v = 0; v < VV; ++v) mx = fmaxf(mx, sS[v * VV + w]);
    float den = 0.f;
    for (int v = 0; v < VV; ++v) den += __expf(sS[v * VV + w] - mx);
    float inv = beta0 / den;
    for (int v = 0; v < VV; ++v)
      sS[v * VV + w] = __expf(sS[v * VV + w] - mx) * inv;
  }
  __syncthreads();

  float alpha0 = alpha[0];
  const float* Ak = A + k * (VV * VV);
  _Float16* out = adyn + ((size_t)(n * COUT + k * 64) << 10);
  // transposed [w][v] padded 32x32, zeros in pads (K pad must be 0)
  for (int i = tid; i < 64 * 1024; i += 256) {
    int c = i >> 10;
    int r = i & 1023;
    int w = r >> 5, v = r & 31;
    float val = 0.f;
    if (v < VV && w < VV)
      val = tanhf(sX1[c * VV + v] - sX2[c * VV + w]) * alpha0 +
            Ak[v * VV + w] + sS[v * VV + w];
    out[((size_t)c << 10) + (w << 5) + v] = (_Float16)val;
  }
}

// ---------------- Kernel 3: pre = relu(bn(W_pre @ x)) as f16 ---------------
// grid (25, 128), block 256 = 8 waves; per block M=192 x Ncols=64, K=64.
// Weights staged via TDM (overlapped with VALU transpose of x tile).
__global__ void __launch_bounds__(256)
k_pre_gemm(const _Float16* __restrict__ x16, const _Float16* __restrict__ Wp16,
           const float* __restrict__ bp, const float* __restrict__ bnp,
           _Float16* __restrict__ pre) {
  __shared__ __align__(16) _Float16 sW[192 * 72];
  __shared__ __align__(16) _Float16 sXT[64 * 72];
  __shared__ float sScale[192], sBias[192];
  int tid = threadIdx.x;
  int n = blockIdx.y;
  int col0 = blockIdx.x * 64;

  weight_stage_issue(Wp16, 64, sW, tid);  // async TDM (or fallback copy)

  for (int o = tid; o < 192; o += 256) {
    float g = bnp[o], b = bnp[192 + o], m = bnp[384 + o], vv = bnp[576 + o];
    float sc = g * rsqrtf(vv + EPSB);
    sScale[o] = sc;
    sBias[o] = b - m * sc + bp[o] * sc;
  }
  const _Float16* xn = x16 + (size_t)n * XSTR + col0;
  for (int i = tid; i < 64 * 32; i += 256) {  // 2 cols per thread
    int c = i >> 5, jp = (i & 31) << 1;
    union { u32 u; _Float16 h[2]; } t;
    t.u = *(const u32*)(xn + c * COLS + jp);
    sXT[jp * 72 + c] = t.h[0];
    sXT[(jp + 1) * 72 + c] = t.h[1];
  }
  weight_stage_wait(tid);
  __syncthreads();

  int wave = tid >> 5, lane = tid & 31;
  _Float16* outn = pre + (size_t)n * OSTR;
  for (int t = wave; t < 48; t += 8) {  // 12 Mtiles x 4 Ntiles
    int mt = t >> 2, nt = t & 3;
    v8f acc = {0.f, 0.f, 0.f, 0.f, 0.f, 0.f, 0.f, 0.f};
#pragma unroll
    for (int k0 = 0; k0 < 64; k0 += 32) {
      v16h a = load_frag_a(sW, 72, mt * 16, k0, lane);
      v16h b = load_frag_b(sXT, 72, k0, nt * 16, lane);
      acc = wmma16(a, b, acc);
    }
    int rb = mt * 16 + ((lane & 16) ? 8 : 0);
    int colL = nt * 16 + (lane & 15);
#pragma unroll
    for (int g = 0; g < 8; ++g) {
      int o = rb + g;
      float val = acc[g] * sScale[o] + sBias[o];
      val = fmaxf(val, 0.f);
      outn[(size_t)o * COLS + col0 + colL] = (_Float16)val;
    }
  }
}

// ---------------- Kernel 4: y[n,o] = pre[n,o](64x25) @ Adyn[n,o](25x25) ----
// Writes y TRANSPOSED: yT[n][col][kc], col stride = 192 halves.
// grid (192, 128), block 64 = 2 waves; wave = Ntile, 4 Mtiles each.
__global__ void __launch_bounds__(64)
k_graphmm(const _Float16* __restrict__ pre, const _Float16* __restrict__ adyn,
          _Float16* __restrict__ yT) {
  __shared__ __align__(16) _Float16 sP[64 * 32];
  int o = blockIdx.x, n = blockIdx.y;
  int tid = threadIdx.x;
  const _Float16* pin = pre + ((size_t)n * COUT + o) * COLS;
  for (int i = tid; i < 64 * 32; i += 64) {
    int t = i >> 5, v = i & 31;
    sP[i] = (v < VV) ? pin[t * VV + v] : (_Float16)0.f;
  }
  __syncthreads();

  int wave = tid >> 5, lane = tid & 31;
  const _Float16* bmat = adyn + ((size_t)(n * COUT + o) << 10);  // [w][v] s=32
  v16h b = load_frag_b(bmat, 32, 0, wave * 16, lane);
  _Float16* yn = yT + (size_t)n * OSTR;
  int colL = wave * 16 + (lane & 15);
#pragma unroll
  for (int mt = 0; mt < 4; ++mt) {
    v8f acc = {0.f, 0.f, 0.f, 0.f, 0.f, 0.f, 0.f, 0.f};
    v16h a = load_frag_a(sP, 32, mt * 16, 0, lane);
    acc = wmma16(a, b, acc);
    if (colL < VV) {
      int rb = mt * 16 + ((lane & 16) ? 8 : 0);
#pragma unroll
      for (int g = 0; g < 8; ++g)
        yn[(size_t)((rb + g) * VV + colL) * COUT + o] = (_Float16)acc[g];
    }
  }
}

// ---------------- Kernel 5: out = relu(bn(W_post@y) + bn(W_down@x)) --------
// grid (50, 128), block 256 = 8 waves; Ncols=32; K segmented:
// segs 0..2 = y chunks of 64 (B frags straight from global yT), seg 3 = x.
__global__ void __launch_bounds__(256)
k_post_gemm(const _Float16* __restrict__ x16, const _Float16* __restrict__ yT,
            const _Float16* __restrict__ Wpost16, const float* __restrict__ bpost,
            const float* __restrict__ bnpost, const _Float16* __restrict__ Wdown16,
            const float* __restrict__ bdown, const float* __restrict__ bndown,
            float* __restrict__ out) {
  __shared__ __align__(16) _Float16 sW[192 * 72];
  __shared__ __align__(16) _Float16 sXT[32 * 72];
  __shared__ float sScP[192], sBiP[192], sScD[192], sBiD[192];
  int tid = threadIdx.x;
  int n = blockIdx.y;
  int col0 = blockIdx.x * 32;

  for (int o = tid; o < 192; o += 256) {
    float g = bnpost[o], b = bnpost[192 + o], m = bnpost[384 + o],
          vv = bnpost[576 + o];
    float sc = g * rsqrtf(vv + EPSB);
    sScP[o] = sc;
    sBiP[o] = b - m * sc + bpost[o] * sc;
    g = bndown[o]; b = bndown[192 + o]; m = bndown[384 + o]; vv = bndown[576 + o];
    sc = g * rsqrtf(vv + EPSB);
    sScD[o] = sc;
    sBiD[o] = b - m * sc + bdown[o] * sc;
  }

  int wave = tid >> 5, lane = tid & 31;
  v8f zero = {0.f, 0.f, 0.f, 0.f, 0.f, 0.f, 0.f, 0.f};
  v8f accP[3], accD[3];
#pragma unroll
  for (int t = 0; t < 3; ++t) { accP[t] = zero; accD[t] = zero; }

  // ---- segs 0..2: W_post chunks (TDM), B from global yT ----
  for (int seg = 0; seg < 3; ++seg) {
    __syncthreads();
    weight_stage_issue(Wpost16 + seg * 64, 192, sW, tid);
    weight_stage_wait(tid);
    __syncthreads();
    const _Float16* ybase =
        yT + (size_t)n * OSTR + (size_t)col0 * COUT + seg * 64;
#pragma unroll
    for (int t = 0; t < 3; ++t) {  // 24 tiles: 12 Mtiles x 2 Ntiles
      int tile = wave + t * 8;
      int mt = tile >> 1, nt = tile & 1;
#pragma unroll
      for (int k0 = 0; k0 < 64; k0 += 32) {
        v16h a = load_frag_a(sW, 72, mt * 16, k0, lane);
        v16h b = load_frag_b(ybase, COUT, k0, nt * 16, lane);
        accP[t] = wmma16(a, b, accP[t]);
      }
    }
  }

  // ---- seg 3: W_down (TDM) + transposed x tile in LDS ----
  __syncthreads();
  weight_stage_issue(Wdown16, 64, sW, tid);
  const _Float16* xn = x16 + (size_t)n * XSTR + col0;
  for (int i = tid; i < 64 * 16; i += 256) {  // 2 cols per thread
    int c = i >> 4, jp = (i & 15) << 1;
    union { u32 u; _Float16 h[2]; } t;
    t.u = *(const u32*)(xn + c * COLS + jp);
    sXT[jp * 72 + c] = t.h[0];
    sXT[(jp + 1) * 72 + c] = t.h[1];
  }
  weight_stage_wait(tid);
  __syncthreads();
#pragma unroll
  for (int t = 0; t < 3; ++t) {
    int tile = wave + t * 8;
    int mt = tile >> 1, nt = tile & 1;
#pragma unroll
    for (int k0 = 0; k0 < 64; k0 += 32) {
      v16h a = load_frag_a(sW, 72, mt * 16, k0, lane);
      v16h b = load_frag_b(sXT, 72, k0, nt * 16, lane);
      accD[t] = wmma16(a, b, accD[t]);
    }
  }

  // ---- epilogue ----
  float* outn = out + (size_t)n * OSTR + col0;
#pragma unroll
  for (int t = 0; t < 3; ++t) {
    int tile = wave + t * 8;
    int mt = tile >> 1, nt = tile & 1;
    int rb = mt * 16 + ((lane & 16) ? 8 : 0);
    int colL = nt * 16 + (lane & 15);
#pragma unroll
    for (int g = 0; g < 8; ++g) {
      int o = rb + g;
      float val = accP[t][g] * sScP[o] + sBiP[o] +
                  accD[t][g] * sScD[o] + sBiD[o];
      outn[(size_t)o * COLS + colL] = fmaxf(val, 0.f);
    }
  }
}

// ---------------- launch ----------------------------------------------------
extern "C" void kernel_launch(void* const* d_in, const int* in_sizes, int n_in,
                              void* d_out, int out_size, void* d_ws,
                              size_t ws_size, hipStream_t stream) {
  const float* x       = (const float*)d_in[0];
  const float* A       = (const float*)d_in[1];
  const float* alpha   = (const float*)d_in[2];
  const float* beta    = (const float*)d_in[3];
  const float* W_pre   = (const float*)d_in[4];
  const float* b_pre   = (const float*)d_in[5];
  const float* bn_pre  = (const float*)d_in[6];
  const float* W1      = (const float*)d_in[7];
  const float* b1      = (const float*)d_in[8];
  const float* W2      = (const float*)d_in[9];
  const float* b2      = (const float*)d_in[10];
  const float* W_post  = (const float*)d_in[11];
  const float* b_post  = (const float*)d_in[12];
  const float* bn_post = (const float*)d_in[13];
  const float* W_down  = (const float*)d_in[14];
  const float* b_down  = (const float*)d_in[15];
  const float* bn_down = (const float*)d_in[16];
  float* out = (float*)d_out;

  char* ws = (char*)d_ws;
  float*    tmp  = (float*)(ws + 0);             //    819,200 B
  _Float16* pre  = (_Float16*)(ws + 819200);     // 78,643,200 B
  _Float16* adyn = (_Float16*)(ws + 79462400);   // 50,331,648 B (32x32 padded)
  _Float16* yT   = (_Float16*)(ws + 129794048);  // 78,643,200 B (col-major)
  _Float16* x16  = (_Float16*)(ws + 208437248);  // 26,214,400 B
  _Float16* w16  = (_Float16*)(ws + 234651648);  //    122,880 B
  _Float16* wpre16  = w16;
  _Float16* wpost16 = w16 + 12288;
  _Float16* wdown16 = w16 + 49152;

  k_cvt_x<<<dim3(12800), 256, 0, stream>>>(x, x16);
  k_cvt_w<<<dim3(240), 256, 0, stream>>>(W_pre, W_post, W_down, w16);
  k_mean<<<dim3(CIN, NB), 32, 0, stream>>>(x, tmp);
  k_adyn<<<dim3(KK, NB), 256, 0, stream>>>(tmp, A, alpha, beta, W1, b1, W2, b2,
                                           adyn);
  k_pre_gemm<<<dim3(25, NB), 256, 0, stream>>>(x16, wpre16, b_pre, bn_pre, pre);
  k_graphmm<<<dim3(COUT, NB), 64, 0, stream>>>(pre, adyn, yT);
  k_post_gemm<<<dim3(50, NB), 256, 0, stream>>>(x16, yT, wpost16, b_post,
                                                bn_post, wdown16, b_down,
                                                bn_down, out);
}
